// FusedMoE_28948079575450
// MI455X (gfx1250) — compile-verified, matrix operand order checked by reading, not run on hardware
//
#include <hip/hip_runtime.h>

#define M_TOK 512
#define K_DIM 4096
#define E_NUM 8
#define N_DIM 14336
#define TOPK  2
#define CAP   (M_TOK * TOPK)   // max routed rows per expert

typedef __attribute__((ext_vector_type(16))) __bf16 v16bf;
typedef __attribute__((ext_vector_type(8)))  __bf16 v8bf;
typedef __attribute__((ext_vector_type(8)))  float  v8f;
typedef __attribute__((ext_vector_type(4)))  unsigned int v4ui;
typedef __attribute__((ext_vector_type(4)))  int    v4i;
typedef __attribute__((ext_vector_type(8)))  int    v8i;

// ---- TDM availability (device pass only; host pass never sees amdgcn builtins)
#if defined(__HIP_DEVICE_COMPILE__) && defined(__gfx1250__) && \
    __has_builtin(__builtin_amdgcn_tensor_load_to_lds) &&      \
    __has_builtin(__builtin_amdgcn_s_wait_tensorcnt)
#define USE_TDM 1
#else
#define USE_TDM 0
#endif

#if USE_TDM && __has_include(<hip/amd_detail/amd_gfx1250_TDM.h>)
#define TDM_6ARG 1
#else
#define TDM_6ARG 0
#endif

// ---- bf16 helpers (round-to-nearest-even via integer ops) ----
__device__ __forceinline__ __bf16 f2bf(float x) {
    unsigned u = __builtin_bit_cast(unsigned, x);
    unsigned r = (u + 0x7fffu + ((u >> 16) & 1u)) >> 16;
    return __builtin_bit_cast(__bf16, (unsigned short)r);
}
__device__ __forceinline__ float bf2f(__bf16 h) {
    unsigned short s = __builtin_bit_cast(unsigned short, h);
    return __builtin_bit_cast(float, ((unsigned)s) << 16);
}

#if USE_TDM
// DMA a 16-row x 64-col fp32 tile (row stride K_DIM) from global into LDS.
// D# group0: count=1 | lds_addr | 57-bit global addr | type=2 ("image").
// D# group1: data_size=4B, tensor_dim0=K_DIM, tensor_dim1=16,
//            tile_dim0=64, tile_dim1=16, tensor_dim0_stride=K_DIM.
__device__ __forceinline__ void tdm_load_tile(const float* gsrc, unsigned ldsOff) {
    unsigned long long ga = (unsigned long long)(size_t)gsrc;
    v4ui g0 = { 1u,                              // count=1 (valid descriptor)
                ldsOff,                          // lds_addr (LDS byte offset)
                (unsigned)ga,                    // global_addr[31:0]
                ((unsigned)(ga >> 32) & 0x01FFFFFFu) | 0x80000000u }; // [56:32]|type=2
    v8i g1 = { (int)0x00020000u,                 // data_size = 4 bytes
               (int)(((unsigned)K_DIM) << 16),   // tensor_dim0[15:0] in [31:16]
               (int)(16u << 16),                 // tensor_dim0[31:16]=0 | tensor_dim1=16
               (int)(64u << 16),                 // tensor_dim1[31:16]=0 | tile_dim0=64
               16,                               // tile_dim1=16 | tile_dim2=0
               K_DIM,                            // tensor_dim0_stride[31:0]
               0, 0 };                           // stride hi / dim1_stride = 0
    v4i gz = { 0, 0, 0, 0 };
#if TDM_6ARG
    v8i gz8 = { 0, 0, 0, 0, 0, 0, 0, 0 };
    __builtin_amdgcn_tensor_load_to_lds(g0, g1, gz, gz, gz8, 0);
#else
    __builtin_amdgcn_tensor_load_to_lds(g0, g1, gz, gz, 0);
#endif
}
#endif

// ---- routing: per-expert lists of flat (token*TOPK + slot) ids ----
__global__ void moe_zero(int* counts) {
    if (threadIdx.x < E_NUM) counts[threadIdx.x] = 0;
}

__global__ void moe_route(const int* __restrict__ topk_ids, int* counts,
                          int* __restrict__ lists) {
    int i = blockIdx.x * blockDim.x + threadIdx.x;
    if (i < CAP) {
        int e = topk_ids[i];
        int pos = atomicAdd(&counts[e], 1);
        lists[e * CAP + pos] = i;
    }
}

// ---- one-time split of A into bf16 hi/lo planes (A reused by 896 N-tiles) ----
__global__ void moe_cvtA(const float* __restrict__ A, __bf16* __restrict__ Ah,
                         __bf16* __restrict__ Al) {
    int i = blockIdx.x * blockDim.x + threadIdx.x;
    if (i < M_TOK * K_DIM) {
        float x = A[i];
        __bf16 h = f2bf(x);
        Ah[i] = h;
        Al[i] = f2bf(x - bf2f(h));
    }
}

// ---- main GEMM: block = (expert e, 16-wide N tile); 8 waves; each wave owns a
// 16-row routed tile; B tile DMA'd by TDM (double buffered), split to bf16
// hi/lo in LDS once per block ----
template <bool PRE>
__launch_bounds__(256)
__global__ void moe_gemm(const float* __restrict__ A,
                         const __bf16* __restrict__ Ah,
                         const __bf16* __restrict__ Al,
                         const float* __restrict__ B,
                         const float* __restrict__ tw,
                         const int* __restrict__ counts,
                         const int* __restrict__ lists,
                         float* __restrict__ out) {
#if USE_TDM
    __shared__ __align__(16) float Bs32[2][16 * 64];  // fp32 staging (TDM dest)
#endif
    __shared__ __bf16 Bh[16 * 72];   // 16 cols x 64 K, padded rows (bank spread)
    __shared__ __bf16 Bl[16 * 72];

    const int e   = blockIdx.y;
    const int n0  = blockIdx.x * 16;
    const int cnt = counts[e];
    if (cnt == 0) return;
    const int rowTiles = (cnt + 15) >> 4;
    const int groups   = (rowTiles + 7) >> 3;

    const int tid  = threadIdx.x;
    const int wave = tid >> 5;
    const int lane = tid & 31;
    const int m    = lane & 15;     // col of B/D, row of A tile
    const int half = lane >> 4;

    const int*   elist = lists + e * CAP;
    const float* Bbase = B + ((size_t)e * N_DIM + n0) * K_DIM;

    // B-staging coords: each thread handles 4 consecutive floats of one row
    const int sn = tid >> 4;          // 0..15 (column of N tile)
    const int sk = (tid & 15) << 2;   // 0..60

    for (int g = 0; g < groups; ++g) {
        const int r0      = (g * 8 + wave) * 16;
        const int rowSlot = (r0 + m < cnt) ? (r0 + m) : (cnt - 1); // clamp
        const int flat    = elist[rowSlot];
        const int token   = flat >> 1;                             // TOPK == 2
        const float*  Arow  = A  + (size_t)token * K_DIM;
        const __bf16* AhRow = Ah + (size_t)token * K_DIM;
        const __bf16* AlRow = Al + (size_t)token * K_DIM;

        v8f c = {};

#if USE_TDM
        if (wave == 0)
            tdm_load_tile(Bbase, (unsigned)(size_t)&Bs32[0][0]);
#endif

        for (int kb = 0; kb < K_DIM; kb += 64) {
#if USE_TDM
            const int buf = (kb >> 6) & 1;
            if (wave == 0) __builtin_amdgcn_s_wait_tensorcnt(0);
            __syncthreads();   // fp32 tile ready AND prev-iter Bh/Bl reads done
            if (wave == 0 && kb + 64 < K_DIM)
                tdm_load_tile(Bbase + kb + 64,
                              (unsigned)(size_t)&Bs32[buf ^ 1][0]);
            {   // split fp32 tile -> bf16 hi/lo (converted once per block)
                const float4 bv = *(const float4*)&Bs32[buf][sn * 64 + sk];
                const int o = sn * 72 + sk;
                #pragma unroll
                for (int j = 0; j < 4; ++j) {
                    float x = (&bv.x)[j];
                    __bf16 h = f2bf(x);
                    Bh[o + j] = h;
                    Bl[o + j] = f2bf(x - bf2f(h));
                }
            }
            __syncthreads();   // bf16 tile ready
#else
            {   // synchronous staging fallback
                const float* src = Bbase + (size_t)sn * K_DIM + kb + sk;
                float4 bv = *(const float4*)src;
                if (kb + 64 < K_DIM)
                    __builtin_prefetch(src + 64, 0, 1);  // global_prefetch_b8
                const int o = sn * 72 + sk;
                #pragma unroll
                for (int j = 0; j < 4; ++j) {
                    float x = (&bv.x)[j];
                    __bf16 h = f2bf(x);
                    Bh[o + j] = h;
                    Bl[o + j] = f2bf(x - bf2f(h));
                }
            }
            __syncthreads();
#endif

            #pragma unroll
            for (int kk = 0; kk < 64; kk += 32) {
                const int kbase = kk + half * 8;

                v16bf ah, al;
                if (PRE) {
                    v8bf a0 = *(const v8bf*)(AhRow + kb + kbase);
                    v8bf a1 = *(const v8bf*)(AhRow + kb + kbase + 16);
                    ah = __builtin_shufflevector(a0, a1, 0,1,2,3,4,5,6,7,
                                                          8,9,10,11,12,13,14,15);
                    v8bf l0 = *(const v8bf*)(AlRow + kb + kbase);
                    v8bf l1 = *(const v8bf*)(AlRow + kb + kbase + 16);
                    al = __builtin_shufflevector(l0, l1, 0,1,2,3,4,5,6,7,
                                                          8,9,10,11,12,13,14,15);
                } else {
                    const float* ap = Arow + kb + kbase;
                    #pragma unroll
                    for (int i = 0; i < 8; ++i) {
                        float x = ap[i];
                        __bf16 h = f2bf(x);
                        ah[i] = h;  al[i] = f2bf(x - bf2f(h));
                        float y = ap[16 + i];
                        __bf16 h2 = f2bf(y);
                        ah[8 + i] = h2;  al[8 + i] = f2bf(y - bf2f(h2));
                    }
                }

                const __bf16* bph = &Bh[m * 72 + kbase];
                const __bf16* bpl = &Bl[m * 72 + kbase];
                v8bf b0 = *(const v8bf*)bph;
                v8bf b1 = *(const v8bf*)(bph + 16);
                v16bf bh = __builtin_shufflevector(b0, b1, 0,1,2,3,4,5,6,7,
                                                            8,9,10,11,12,13,14,15);
                v8bf c0 = *(const v8bf*)bpl;
                v8bf c1 = *(const v8bf*)(bpl + 16);
                v16bf bl = __builtin_shufflevector(c0, c1, 0,1,2,3,4,5,6,7,
                                                            8,9,10,11,12,13,14,15);

                // D = ah*bh + ah*bl + al*bh + C  (near-fp32 accuracy)
                c = __builtin_amdgcn_wmma_f32_16x16x32_bf16(false, ah, false, bh,
                                                            (short)0, c, false, false);
                c = __builtin_amdgcn_wmma_f32_16x16x32_bf16(false, ah, false, bl,
                                                            (short)0, c, false, false);
                c = __builtin_amdgcn_wmma_f32_16x16x32_bf16(false, al, false, bh,
                                                            (short)0, c, false, false);
            }
#if !USE_TDM
            __syncthreads();
#endif
        }

        // --- store: C/D layout: VGPR v -> row half*8+v, col = lane%16 ---
        #pragma unroll
        for (int v = 0; v < 8; ++v) {
            const int tr = half * 8 + v;
            if (r0 + tr < cnt) {
                const int f = elist[r0 + tr];
                out[(size_t)f * N_DIM + n0 + m] = c[v] * tw[f];
            }
        }
    }
}

extern "C" void kernel_launch(void* const* d_in, const int* in_sizes, int n_in,
                              void* d_out, int out_size, void* d_ws, size_t ws_size,
                              hipStream_t stream) {
    (void)in_sizes; (void)n_in; (void)out_size;
    const float* A   = (const float*)d_in[0];
    const float* B   = (const float*)d_in[1];
    const float* tw  = (const float*)d_in[2];
    const int*   ids = (const int*)d_in[3];
    float* out = (float*)d_out;

    int* counts = (int*)d_ws;            // 64 ints (256 B)
    int* lists  = counts + 64;           // E*CAP ints (32 KB)
    const size_t ABF_OFF = 64 * 1024;
    __bf16* Ah = (__bf16*)((char*)d_ws + ABF_OFF);
    __bf16* Al = Ah + (size_t)M_TOK * K_DIM;
    const size_t need = ABF_OFF + (size_t)M_TOK * K_DIM * 2 * sizeof(__bf16);

    moe_zero<<<1, 32, 0, stream>>>(counts);
    moe_route<<<(CAP + 255) / 256, 256, 0, stream>>>(ids, counts, lists);

    dim3 grid(N_DIM / 16, E_NUM);
    if (ws_size >= need) {
        moe_cvtA<<<(M_TOK * K_DIM + 255) / 256, 256, 0, stream>>>(A, Ah, Al);
        moe_gemm<true><<<grid, 256, 0, stream>>>(A, Ah, Al, B, tw, counts, lists, out);
    } else {
        moe_gemm<false><<<grid, 256, 0, stream>>>(A, Ah, Al, B, tw, counts, lists, out);
    }
}
